// NonLocal2D_75041668595894
// MI455X (gfx1250) — matrix operand order repaired and checked by reading, not compiled, not run on hardware
//
#include <hip/hip_runtime.h>
#include <math.h>

typedef __attribute__((ext_vector_type(16))) _Float16 v16h;
typedef __attribute__((ext_vector_type(8)))  float    v8f;
typedef _Float16 h16;

static constexpr int C_IN    = 256;
static constexpr int C_INTER = 128;
static constexpr int BATCH   = 8;
static constexpr int LPIX    = 4096; // H*W

__device__ __forceinline__ v8f vzero() {
  v8f z = {0.f,0.f,0.f,0.f,0.f,0.f,0.f,0.f};
  return z;
}

// WMMA f32 = f16 x f16 + f32, 16x16x32
__device__ __forceinline__ v8f wmma16(v16h a, v16h b, v8f c) {
  return __builtin_amdgcn_wmma_f32_16x16x32_f16(
      /*neg_a=*/false, a, /*neg_b=*/false, b,
      /*c_mod=*/(short)0, c, /*reuse_a=*/false, /*reuse_b=*/false);
}

// A-fragment (16x32, 16-bit) from row-major [rows, ld] f16 source.
// ISA layout: lane ln holds row M=ln; hi selects K offset:
//   elems 0..7  -> K = hi*8 + e ; elems 8..15 -> K = 16 + hi*8 + (e-8)
__device__ __forceinline__ v16h load_a16(const h16* base, int ld, int row,
                                         int col0, int hi) {
  const h16* p0 = base + (size_t)row * ld + col0 + hi * 8;
  const h16* p1 = p0 + 16;
  v16h f;
#pragma unroll
  for (int e = 0; e < 8; ++e) { f[e] = p0[e]; f[8 + e] = p1[e]; }
  return f;
}

// B-fragment (32x16, 16-bit) given B^T stored row-major as [N, K]:
// lane ln holds column n=ln; elems 0..7 -> K = hi*8 + e ; elems 8..15 -> K = 16 + hi*8 + (e-8).
__device__ __forceinline__ v16h load_b16(const h16* bt, int ld, int ncol,
                                         int k0, int hi) {
  const h16* p = bt + (size_t)ncol * ld + k0 + hi * 8;
  v16h f;
#pragma unroll
  for (int e = 0; e < 8; ++e) { f[e] = p[e]; f[8 + e] = p[16 + e]; }
  return f;
}

// ---------------- Kernel 0: convert weights f32 -> f16 ----------------
// dst layout (halves): [Wg][Wt][Wp][Wo], each 128*256
__global__ void __launch_bounds__(256)
cvt_weights_kernel(const float* __restrict__ Wg, const float* __restrict__ Wt,
                   const float* __restrict__ Wp, const float* __restrict__ Wo,
                   h16* __restrict__ dst) {
  const int N = C_INTER * C_IN;
  int i = blockIdx.x * blockDim.x + threadIdx.x;
  if (i >= 4 * N) return;
  const float* src;
  int off = i;
  if (i < N)            { src = Wg; }
  else if (i < 2 * N)   { src = Wt; off -= N; }
  else if (i < 3 * N)   { src = Wp; off -= 2 * N; }
  else                  { src = Wo; off -= 3 * N; }
  dst[i] = (h16)src[off];
}

// ---------------- Kernel 1: QKV projection ----------------
// grid (LPIX/64, 3, B), block 128 (4 waves). Wave computes 16(L) x 128(d).
// proj 0: V = Wg x + bg, stored TRANSPOSED [128, L]
// proj 1: Q = Wt x + bt, stored [L, 128]
// proj 2: K = Wp x + bp, stored [L, 128]
__global__ void __launch_bounds__(128)
proj_kernel(const float* __restrict__ x, const h16* __restrict__ Wh,
            const float* __restrict__ bg, const float* __restrict__ bt,
            const float* __restrict__ bp, h16* __restrict__ Q,
            h16* __restrict__ K, h16* __restrict__ Vt) {
  const int proj  = blockIdx.y;
  const int b     = blockIdx.z;
  const int l0blk = blockIdx.x * 64;
  const int tid   = threadIdx.x;
  const int wave  = tid >> 5;
  const int lane  = tid & 31;
  const int ln = lane & 15, hi = lane >> 4;

  __shared__ h16 xs[64][40]; // [l][c] staged tile, padded

  const float* xb   = x + (size_t)b * C_IN * LPIX;
  const h16*   W    = Wh + (size_t)proj * C_INTER * C_IN;
  const float* bias = (proj == 0) ? bg : (proj == 1) ? bt : bp;

  v8f acc[8];
#pragma unroll
  for (int j = 0; j < 8; ++j) acc[j] = vzero();

  for (int c0 = 0; c0 < C_IN; c0 += 32) {
    __syncthreads();
    // Stage x[b, c0..c0+32, l0..l0+64] transposed into xs[l][c] (f16),
    // coalesced on global (contiguous in l).
#pragma unroll
    for (int it = 0; it < 16; ++it) {
      int idx = tid + it * 128;     // 0..2047
      int c = idx >> 6;             // 0..31
      int l = idx & 63;             // 0..63
      xs[l][c] = (h16)xb[(size_t)(c0 + c) * LPIX + l0blk + l];
    }
    __syncthreads();

    const int lrow = wave * 16 + ln;
    v16h afrag = load_a16(&xs[0][0], 40, lrow, 0, hi);
#pragma unroll
    for (int j = 0; j < 8; ++j) {
      v16h bfrag = load_b16(W, C_IN, j * 16 + ln, c0, hi); // W[d, c]
      acc[j] = wmma16(afrag, bfrag, acc[j]);
    }
  }

  const int l0 = l0blk + wave * 16;
#pragma unroll
  for (int j = 0; j < 8; ++j) {
    float bj = bias[j * 16 + ln];
    if (proj == 0) {
      h16* dst = Vt + (size_t)b * C_INTER * LPIX +
                 (size_t)(j * 16 + ln) * LPIX + l0 + hi * 8;
#pragma unroll
      for (int i = 0; i < 8; ++i) dst[i] = (h16)(acc[j][i] + bj);
    } else {
      h16* dst = ((proj == 1) ? Q : K) + (size_t)b * LPIX * C_INTER;
#pragma unroll
      for (int i = 0; i < 8; ++i)
        dst[(size_t)(l0 + i + 8 * hi) * C_INTER + j * 16 + ln] =
            (h16)(acc[j][i] + bj);
    }
  }
}

// ---------------- Kernel 2: flash attention, LDS-shared K/V ----------------
// grid (LPIX/128, B), block 256 (8 waves). Each wave owns 16 query rows;
// the block cooperatively stages each 32-key K/V tile in LDS once,
// cutting K/V L2 traffic 8x vs per-wave streaming.
__global__ void __launch_bounds__(256)
attn_kernel(const h16* __restrict__ Q, const h16* __restrict__ K,
            const h16* __restrict__ Vt, h16* __restrict__ Y) {
  const int b    = blockIdx.y;
  const int tid  = threadIdx.x;
  const int wave = tid >> 5;       // 0..7
  const int lane = tid & 31;
  const int ln = lane & 15, hi = lane >> 4;
  const int q0 = blockIdx.x * 128 + wave * 16;

  const h16* Qb = Q + (size_t)b * LPIX * C_INTER;
  const h16* Kb = K + (size_t)b * LPIX * C_INTER;
  const h16* Vb = Vt + (size_t)b * C_INTER * LPIX;

  // Padded rows: K row = 136 halves (272B = 68 DW -> +4 banks/row),
  // V/P row = 40 halves (80B = 20 DW -> 16 distinct bank offsets).
  __shared__ __align__(16) h16 Ks[32][136];   // [key][d]
  __shared__ __align__(16) h16 Vs[128][40];   // [d][key]
  __shared__ __align__(16) h16 Pbuf[8][16 * 40];
  h16* P = Pbuf[wave];

  // Q fragments: 4 k-chunks of 32 over d=128, reused for all key tiles
  v16h qf[4];
#pragma unroll
  for (int kc = 0; kc < 4; ++kc)
    qf[kc] = load_a16(Qb, C_INTER, q0 + ln, kc * 32, hi);

  v8f o[8];
#pragma unroll
  for (int j = 0; j < 8; ++j) o[j] = vzero();
  float mrow[8], lrow[8];
#pragma unroll
  for (int i = 0; i < 8; ++i) { mrow[i] = -__builtin_inff(); lrow[i] = 0.f; }

  for (int k0 = 0; k0 < LPIX; k0 += 32) {
    __syncthreads();
    // Cooperative stage: K tile 32x128 and V tile 128x32 (16B per thread/pass)
#pragma unroll
    for (int p = 0; p < 2; ++p) {
      int h = (p * 256 + tid) * 8;          // half-element index
      int key = h >> 7, c = h & 127;        // K tile coords
      *(uint4*)&Ks[key][c] =
          *(const uint4*)(Kb + (size_t)(k0 + key) * C_INTER + c);
      int d = h >> 5, kk = h & 31;          // V tile coords
      *(uint4*)&Vs[d][kk] =
          *(const uint4*)(Vb + (size_t)d * LPIX + k0 + kk);
    }
    __syncthreads();
    if (k0 + 32 < LPIX) {
      __builtin_prefetch(Kb + (size_t)(k0 + 32) * C_INTER + (tid & 63) * 64,
                         0, 3);
    }

    // S tile: 16 q-rows x 32 keys = two 16x16 accumulators, K from LDS
    v8f sa = vzero(), sb = vzero();
#pragma unroll
    for (int kc = 0; kc < 4; ++kc) {
      v16h kfa = load_b16(&Ks[0][0], 136, ln, kc * 32, hi);
      v16h kfb = load_b16(&Ks[0][0], 136, 16 + ln, kc * 32, hi);
      sa = wmma16(qf[kc], kfa, sa);
      sb = wmma16(qf[kc], kfb, sb);
    }

    // Online softmax per accumulator row i (row M = i + 8*hi)
#pragma unroll
    for (int i = 0; i < 8; ++i) {
      float smax = fmaxf(sa[i], sb[i]);
#pragma unroll
      for (int off = 1; off < 16; off <<= 1)
        smax = fmaxf(smax, __shfl_xor(smax, off, 32));
      float mnew  = fmaxf(mrow[i], smax);
      float scale = __expf(mrow[i] - mnew);
      float ea = __expf(sa[i] - mnew);
      float eb = __expf(sb[i] - mnew);
      float rs = ea + eb;
#pragma unroll
      for (int off = 1; off < 16; off <<= 1)
        rs += __shfl_xor(rs, off, 32);
      lrow[i] = lrow[i] * scale + rs;
      mrow[i] = mnew;
#pragma unroll
      for (int j = 0; j < 8; ++j) o[j][i] *= scale;
      // stash P in LDS for C-layout -> A-layout conversion
      int prow = i + 8 * hi;
      P[prow * 40 + ln]      = (h16)ea;
      P[prow * 40 + 16 + ln] = (h16)eb;
    }

    // O += P(16x32) @ V(32x128), V from LDS
    v16h pf = load_a16(P, 40, ln, 0, hi);
#pragma unroll
    for (int j = 0; j < 8; ++j) {
      v16h vf = load_b16(&Vs[0][0], 40, j * 16 + ln, 0, hi);
      o[j] = wmma16(pf, vf, o[j]);
    }
  }

  // normalize and store Y [L, 128] f16
  float inv[8];
#pragma unroll
  for (int i = 0; i < 8; ++i) inv[i] = 1.0f / lrow[i];
  h16* Yb = Y + ((size_t)b * LPIX + q0) * C_INTER;
#pragma unroll
  for (int j = 0; j < 8; ++j) {
#pragma unroll
    for (int i = 0; i < 8; ++i) {
      Yb[(size_t)(i + 8 * hi) * C_INTER + j * 16 + ln] =
          (h16)(o[j][i] * inv[i]);
    }
  }
}

// ---------------- Kernel 3: output projection + bias + residual ----------------
// grid (LPIX/64, B), block 128. Wave: 16(L) x 256(C) via 16 column tiles.
__global__ void __launch_bounds__(128)
out_kernel(const h16* __restrict__ Y, const h16* __restrict__ WoH,
           const float* __restrict__ bo, const float* __restrict__ x,
           float* __restrict__ out) {
  const int b    = blockIdx.y;
  const int tid  = threadIdx.x;
  const int wave = tid >> 5;
  const int lane = tid & 31;
  const int ln = lane & 15, hi = lane >> 4;
  const int l0 = blockIdx.x * 64 + wave * 16;

  const h16*   Yb = Y + (size_t)b * LPIX * C_INTER;
  const float* xb = x + (size_t)b * C_IN * LPIX;
  float*       ob = out + (size_t)b * C_IN * LPIX;

  v16h af[4];
#pragma unroll
  for (int kc = 0; kc < 4; ++kc)
    af[kc] = load_a16(Yb, C_INTER, l0 + ln, kc * 32, hi);

  for (int ct = 0; ct < 16; ++ct) {
    v8f acc = vzero();
#pragma unroll
    for (int kc = 0; kc < 4; ++kc) {
      v16h bf = load_b16(WoH, C_INTER, ct * 16 + ln, kc * 32, hi); // Wo[co,d]
      acc = wmma16(af[kc], bf, acc);
    }
    int co = ct * 16 + ln;
    float bias = bo[co];
    const float* xp = xb + (size_t)co * LPIX + l0 + hi * 8;
    float*       op = ob + (size_t)co * LPIX + l0 + hi * 8;
#pragma unroll
    for (int i = 0; i < 8; ++i) op[i] = acc[i] + bias + xp[i];
  }
}

// ---------------- Launcher ----------------
extern "C" void kernel_launch(void* const* d_in, const int* in_sizes, int n_in,
                              void* d_out, int out_size, void* d_ws,
                              size_t ws_size, hipStream_t stream) {
  const float* x  = (const float*)d_in[0];
  const float* Wg = (const float*)d_in[1];
  const float* bg = (const float*)d_in[2];
  const float* Wt = (const float*)d_in[3];
  const float* bt = (const float*)d_in[4];
  const float* Wp = (const float*)d_in[5];
  const float* bp = (const float*)d_in[6];
  const float* Wo = (const float*)d_in[7];
  const float* bo = (const float*)d_in[8];
  float* out = (float*)d_out;

  const size_t WSZ = (size_t)C_INTER * C_IN;          // 32768 halves per weight
  const size_t QSZ = (size_t)BATCH * LPIX * C_INTER;  // per Q/K/Vt/Y buffer

  h16* Wh  = (h16*)d_ws;        // [Wg][Wt][Wp] f16
  h16* WoH = Wh + 3 * WSZ;      // Wo f16
  h16* Q   = WoH + WSZ;
  h16* K   = Q + QSZ;
  h16* Vt  = K + QSZ;
  h16* Y   = Vt + QSZ;

  (void)in_sizes; (void)n_in; (void)out_size; (void)ws_size;

  cvt_weights_kernel<<<(int)((4 * WSZ + 255) / 256), 256, 0, stream>>>(
      Wg, Wt, Wp, Wo, Wh);

  dim3 g1(LPIX / 64, 3, BATCH);
  proj_kernel<<<g1, 128, 0, stream>>>(x, Wh, bg, bt, bp, Q, K, Vt);

  dim3 g2(LPIX / 128, BATCH);
  attn_kernel<<<g2, 256, 0, stream>>>(Q, K, Vt, Y);

  dim3 g3(LPIX / 64, BATCH);
  out_kernel<<<g3, 128, 0, stream>>>(Y, WoH, bo, x, out);
}
